// PortfolioPerformer_8358006358284
// MI455X (gfx1250) — matrix operand order repaired and checked by reading, not compiled, not run on hardware
//
#include <hip/hip_runtime.h>
#include <hip/hip_bf16.h>
#include <math.h>

typedef _Float16 h16;
typedef __attribute__((ext_vector_type(16))) _Float16 v16h;
typedef __attribute__((ext_vector_type(8)))  _Float16 v8h;
typedef __attribute__((ext_vector_type(8)))  float    v8f;

#define SEQ    512           // B*A sequences
#define LSEQ   512           // tokens per sequence
#define DMODEL 64
#define NHEAD  4
#define DHEAD  16
#define MFEAT  44
#define FFD    256
#define EPSK   1e-4f
#define RATIO  0.15075567228888181f   /* 44^-0.5 */

// f16 transposed weight pack per layer (offsets in halfs)
#define WL  49152
#define WQT 0
#define WKT 4096
#define WVT 8192
#define WOT 12288
#define W1T 16384
#define W2T 32768

// ---------------- WMMA helpers (CDNA5 16x16x32 f16 -> f32) ----------------

__device__ __forceinline__ v8f wmma_f16(v16h a, v16h b, v8f c) {
  // D = A(16x32) * B(32x16) + C ; emits v_wmma_f32_16x16x32_f16
  return __builtin_amdgcn_wmma_f32_16x16x32_f16(false, a, false, b, (short)0, c,
                                                false, false);
}

// A fragment: row-major A[row][k], lda in halfs.
// lane<16: row=lane, K = kb..kb+7 and kb+16..kb+23 with kb=0; lane>=16: kb=8.
__device__ __forceinline__ v16h load_frag_a(const h16* A, int lda, int row0, int k0) {
  int lane = threadIdx.x & 31;
  int r  = lane & 15;
  int kb = (lane >> 4) * 8;
  const h16* p = A + (size_t)(row0 + r) * lda + k0 + kb;
  v8h lo = *(const v8h*)(p);
  v8h hi = *(const v8h*)(p + 16);
  v16h a;
#pragma unroll
  for (int i = 0; i < 8; ++i) { a[i] = lo[i]; a[i + 8] = hi[i]; }
  return a;
}

// B fragment from pre-transposed Bt[n][k] (row-major, ldb in halfs):
// lane<16: col=lane, K=0..7,16..23 ; lane>=16: col=lane-16, K=8..15,24..31 (within k0 step).
__device__ __forceinline__ v16h load_frag_b(const h16* Bt, int ldb, int n0, int k0) {
  int lane = threadIdx.x & 31;
  int c  = lane & 15;
  int kb = (lane >> 4) * 8;
  const h16* p = Bt + (size_t)(n0 + c) * ldb + k0 + kb;
  v8h lo = *(const v8h*)(p);
  v8h hi = *(const v8h*)(p + 16);
  v16h b;
#pragma unroll
  for (int i = 0; i < 8; ++i) { b[i] = lo[i]; b[i + 8] = hi[i]; }
  return b;
}

__device__ __forceinline__ float gelu_tanh(float x) {
  float x3 = x * x * x;
  return 0.5f * x * (1.f + tanhf(0.7978845608028654f * (x + 0.044715f * x3)));
}

// ---------------- kernels ----------------

// Convert all layer weights to f16, transposed [N][K] for B-fragment loads.
__global__ void k_prep_weights(const float* __restrict__ wq, const float* __restrict__ wk,
                               const float* __restrict__ wv, const float* __restrict__ wo,
                               const float* __restrict__ w1, const float* __restrict__ w2,
                               h16* __restrict__ wT) {
  for (int i = blockIdx.x * blockDim.x + threadIdx.x; i < 2 * WL;
       i += gridDim.x * blockDim.x) {
    int l = i / WL, j = i % WL;
    float v;
    if (j < 16384) {                       // wq|wk|wv|wo : [64][64] -> [n][k]
      int mat = j >> 12, r = j & 4095, n = r >> 6, k = r & 63;
      const float* src = (mat == 0) ? wq : (mat == 1) ? wk : (mat == 2) ? wv : wo;
      v = src[l * 4096 + k * 64 + n];
    } else if (j < 32768) {                // w1 : [64][256] -> [256][64]
      int r = j - 16384, n = r >> 6, k = r & 63;
      v = w1[l * 16384 + k * 256 + n];
    } else {                               // w2 : [256][64] -> [64][256]
      int r = j - 32768, n = r >> 8, k = r & 255;
      v = w2[l * 16384 + k * 64 + n];
    }
    wT[i] = (h16)v;
  }
}

// h[seq][t][d] = x0*w_in[0][d] + x1*w_in[1][d] + b_in[d]
__global__ void k_input_proj(const float* __restrict__ x, const float* __restrict__ w_in,
                             const float* __restrict__ b_in, float* __restrict__ h) {
  int seq = blockIdx.x;
  for (int idx = threadIdx.x; idx < LSEQ * DMODEL; idx += blockDim.x) {
    int t = idx >> 6, d = idx & 63;
    size_t xo = ((size_t)seq * LSEQ + t) * 2;
    float x0 = x[xo], x1 = x[xo + 1];
    h[((size_t)seq * LSEQ + t) * DMODEL + d] = x0 * w_in[d] + x1 * w_in[64 + d] + b_in[d];
  }
}

// LayerNorm over last dim (64), one wave per row, writes f16.
__global__ void k_layernorm(const float* __restrict__ h, const float* __restrict__ g,
                            const float* __restrict__ b, h16* __restrict__ hn) {
  int lane = threadIdx.x & 31;
  int wid  = threadIdx.x >> 5;
  size_t row = (size_t)blockIdx.x * 8 + wid;
  const float* p = h + row * DMODEL;
  float e0 = p[lane], e1 = p[lane + 32];
  float s = e0 + e1;
#pragma unroll
  for (int off = 16; off; off >>= 1) s += __shfl_xor(s, off, 32);
  float mean = s * (1.f / 64.f);
  float d0 = e0 - mean, d1 = e1 - mean;
  float q = d0 * d0 + d1 * d1;
#pragma unroll
  for (int off = 16; off; off >>= 1) q += __shfl_xor(q, off, 32);
  float rs = rsqrtf(q * (1.f / 64.f) + 1e-5f);
  hn[row * DMODEL + lane]      = (h16)(d0 * rs * g[lane]      + b[lane]);
  hn[row * DMODEL + lane + 32] = (h16)(d1 * rs * g[lane + 32] + b[lane + 32]);
}

// Fused FAVOR+ attention for one sequence per workgroup (128 threads = 4 waves).
// Phase 1: V and K via WMMA; u = xs@pm^T - diag stored f16 in LDS; per-seq max.
// Phase 2: per-head wave runs the causal scan with on-the-fly q/qp (WMMA q tiles).
// Single occupancy per WGP (256 KB LDS) -> let the compiler use VGPRs freely.
__global__ void __launch_bounds__(128, 1)
k_attention(const h16* __restrict__ hn_all, const h16* __restrict__ wT,
            const float* __restrict__ pm, h16* __restrict__ ob_all) {
  extern __shared__ char smem[];
  h16*   kpb  = (h16*)smem;                      // [4][512][44] f16  (180224 B)
  h16*   vb   = (h16*)(smem + 180224);           // [512][64]   f16  ( 65536 B)
  float* kst  = (float*)(smem + 245760);         // [4][16][17] f32  (  4352 B)
  float* qpst = (float*)(smem + 250112);         // [4][16][44] f32  ( 11264 B)
  float* red  = (float*)(smem + 261376);         // [128]       f32  (   512 B)

  const int tid  = threadIdx.x;
  const int lane = tid & 31;
  const int wid  = tid >> 5;                 // wave 0..3
  const int seq  = blockIdx.x;
  const h16* hn  = hn_all + (size_t)seq * LSEQ * DMODEL;
  h16*       ob  = ob_all + (size_t)seq * LSEQ * DMODEL;
  const h16* wqT = wT + WQT;
  const h16* wkT = wT + WKT;
  const h16* wvT = wT + WVT;
  float* kst_w   = kst  + wid * (16 * 17);
  float* qpst_w  = qpst + wid * (16 * MFEAT);

  // ---- phase 1a: V = hn @ Wv -> LDS ----
  for (int ti = wid; ti < 128; ti += 4) {
    int rc = ti >> 2, nt = ti & 3;
    int row0 = rc * 16, n0 = nt * 16;
    v8f acc = {};
    acc = wmma_f16(load_frag_a(hn, DMODEL, row0, 0),  load_frag_b(wvT, DMODEL, n0, 0),  acc);
    acc = wmma_f16(load_frag_a(hn, DMODEL, row0, 32), load_frag_b(wvT, DMODEL, n0, 32), acc);
    int part = lane >> 4, c = lane & 15;
#pragma unroll
    for (int r = 0; r < 8; ++r)
      vb[(row0 + part * 8 + r) * DMODEL + n0 + c] = (h16)acc[r];
  }
  __syncthreads();

  // ---- phase 1b: K tiles -> u - diag (f16 in kpb), track max(u) ----
  float mx = -1e30f;
  for (int ti = wid; ti < 128; ti += 4) {
    int rc = ti >> 2, head = ti & 3;
    int row0 = rc * 16, n0 = head * 16;
    v8f acc = {};
    acc = wmma_f16(load_frag_a(hn, DMODEL, row0, 0),  load_frag_b(wkT, DMODEL, n0, 0),  acc);
    acc = wmma_f16(load_frag_a(hn, DMODEL, row0, 32), load_frag_b(wkT, DMODEL, n0, 32), acc);
    int part = lane >> 4, c = lane & 15;
#pragma unroll
    for (int r = 0; r < 8; ++r)
      kst_w[(part * 8 + r) * 17 + c] = acc[r];
    __syncthreads();
    int row   = lane & 15;
    int mbase = (lane >> 4) * 22;
    float ksq = 0.f;
#pragma unroll
    for (int dd = 0; dd < 16; ++dd) { float kk = kst_w[row * 17 + dd]; ksq += kk * kk; }
    float diag = 0.125f * ksq;                 // 0.5 * sum((0.5k)^2)
    int rowg = row0 + row;
#pragma unroll
    for (int mi = 0; mi < 22; ++mi) {
      int m = mbase + mi;
      float dot = 0.f;
#pragma unroll
      for (int dd = 0; dd < 16; ++dd) dot += kst_w[row * 17 + dd] * pm[m * 16 + dd];
      float u = 0.5f * dot;                    // d^-0.25 = 0.5 for d=16
      mx = fmaxf(mx, u);
      kpb[(head * LSEQ + rowg) * MFEAT + m] = (h16)(u - diag);
    }
    __syncthreads();
  }

  // ---- block max reduce (key stabilizer, per-sequence approximation) ----
  red[tid] = mx;
  __syncthreads();
  for (int s = 64; s > 0; s >>= 1) {
    if (tid < s) red[tid] = fmaxf(red[tid], red[tid + s]);
    __syncthreads();
  }
  float gmax = red[0];

  // ---- phase 1c: kp = (exp(u - diag - gmax) + eps) * ratio ----
  for (int i = tid; i < NHEAD * LSEQ * MFEAT; i += 128) {
    float w = (float)kpb[i];
    kpb[i] = (h16)((expf(w - gmax) + EPSK) * RATIO);
  }
  __syncthreads();

  // ---- phase 2: causal scan, wave = head ----
  const int head  = wid;
  const int d     = lane & 15;
  const int part  = lane >> 4;
  const int mbase = part * 22;
  float S[22], z[22];
#pragma unroll
  for (int i = 0; i < 22; ++i) { S[i] = 0.f; z[i] = 0.f; }

  for (int t0 = 0; t0 < LSEQ; t0 += 16) {
    if (t0 + 16 < LSEQ) __builtin_prefetch(hn + (size_t)(t0 + 16) * DMODEL, 0, 1);
    // q tile for this chunk (16 rows x this head's 16 dims)
    v8f acc = {};
    acc = wmma_f16(load_frag_a(hn, DMODEL, t0, 0),  load_frag_b(wqT, DMODEL, head * 16, 0),  acc);
    acc = wmma_f16(load_frag_a(hn, DMODEL, t0, 32), load_frag_b(wqT, DMODEL, head * 16, 32), acc);
    {
      int c = lane & 15;
#pragma unroll
      for (int r = 0; r < 8; ++r) kst_w[(part * 8 + r) * 17 + c] = acc[r];
    }
    __syncthreads();
    // qp for the 16 rows (per-row max stabilizer). Raw u staged in LDS (each
    // lane re-reads only its own entries -> per-wave DS ordering suffices).
    {
      int row = lane & 15;
      float qsq = 0.f;
#pragma unroll
      for (int dd = 0; dd < 16; ++dd) { float qq = kst_w[row * 17 + dd]; qsq += qq * qq; }
      float diag = 0.125f * qsq;
      float lmax = -1e30f;
#pragma unroll
      for (int mi = 0; mi < 22; ++mi) {
        float dot = 0.f;
#pragma unroll
        for (int dd = 0; dd < 16; ++dd) dot += kst_w[row * 17 + dd] * pm[(mbase + mi) * 16 + dd];
        float u = 0.5f * dot;
        lmax = fmaxf(lmax, u);
        qpst_w[row * MFEAT + mbase + mi] = u;
      }
      lmax = fmaxf(lmax, __shfl_xor(lmax, 16, 32));   // combine both m-halves
#pragma unroll
      for (int mi = 0; mi < 22; ++mi) {
        float u = qpst_w[row * MFEAT + mbase + mi];
        qpst_w[row * MFEAT + mbase + mi] = (expf(u - diag - lmax) + EPSK) * RATIO;
      }
    }
    __syncthreads();
    // 16 sequential prefix-sum steps
    for (int i = 0; i < 16; ++i) {
      int t = t0 + i;
      float vt = (float)vb[t * DMODEL + head * 16 + d];
      float num = 0.f, den = 0.f;
#pragma unroll
      for (int mi = 0; mi < 22; ++mi) {
        float kt = (float)kpb[(head * LSEQ + t) * MFEAT + mbase + mi];
        S[mi] += kt * vt;
        z[mi] += kt;
        float qt = qpst_w[i * MFEAT + mbase + mi];
        num += qt * S[mi];
        den += qt * z[mi];
      }
      num += __shfl_xor(num, 16, 32);
      den += __shfl_xor(den, 16, 32);
      float o = num / (den + 1e-6f);
      if (part == 0) ob[(size_t)t * DMODEL + head * 16 + d] = (h16)o;
    }
    __syncthreads();
  }
}

// h += o @ Wo + bo   (16-row x 64-col tile per block, one N-tile per wave)
__global__ void k_wo_residual(const h16* __restrict__ ob_all, const h16* __restrict__ woT,
                              const float* __restrict__ bo, float* __restrict__ h_all) {
  int seq = blockIdx.x >> 5, rc = blockIdx.x & 31;
  int lane = threadIdx.x & 31, wid = threadIdx.x >> 5;
  const h16* A = ob_all + (size_t)seq * LSEQ * DMODEL;
  float* hmat  = h_all  + (size_t)seq * LSEQ * DMODEL;
  int row0 = rc * 16, n0 = wid * 16;
  v8f acc = {};
  acc = wmma_f16(load_frag_a(A, DMODEL, row0, 0),  load_frag_b(woT, DMODEL, n0, 0),  acc);
  acc = wmma_f16(load_frag_a(A, DMODEL, row0, 32), load_frag_b(woT, DMODEL, n0, 32), acc);
  int part = lane >> 4, c = lane & 15;
#pragma unroll
  for (int r = 0; r < 8; ++r) {
    int row = row0 + part * 8 + r, col = n0 + c;
    hmat[(size_t)row * DMODEL + col] += acc[r] + bo[col];
  }
}

// h += gelu(hn @ W1 + b1) @ W2 + b2   (fused, mid kept in LDS as f16)
__global__ void k_ffn(const h16* __restrict__ hn_all, const h16* __restrict__ w1T,
                      const h16* __restrict__ w2T, const float* __restrict__ b1,
                      const float* __restrict__ b2, float* __restrict__ h_all) {
  __shared__ __align__(16) h16 mid[16 * 264];   // [16 rows][256+pad]
  int seq = blockIdx.x >> 5, rc = blockIdx.x & 31;
  int lane = threadIdx.x & 31, wid = threadIdx.x >> 5;
  int part = lane >> 4, c = lane & 15;
  const h16* A = hn_all + (size_t)seq * LSEQ * DMODEL;
  float* hmat  = h_all  + (size_t)seq * LSEQ * DMODEL;
  int row0 = rc * 16;
  // phase A: mid = gelu(hn @ W1 + b1)
  for (int nt = wid; nt < 16; nt += 4) {
    int n0 = nt * 16;
    v8f acc = {};
    acc = wmma_f16(load_frag_a(A, DMODEL, row0, 0),  load_frag_b(w1T, DMODEL, n0, 0),  acc);
    acc = wmma_f16(load_frag_a(A, DMODEL, row0, 32), load_frag_b(w1T, DMODEL, n0, 32), acc);
#pragma unroll
    for (int r = 0; r < 8; ++r) {
      int col = n0 + c;
      mid[(part * 8 + r) * 264 + col] = (h16)gelu_tanh(acc[r] + b1[col]);
    }
  }
  __syncthreads();
  // phase B: h += mid @ W2 + b2  (K=256 -> 8 WMMA, A-fragments from LDS)
  {
    int n0 = wid * 16;
    v8f acc = {};
#pragma unroll
    for (int k0 = 0; k0 < FFD; k0 += 32)
      acc = wmma_f16(load_frag_a(mid, 264, 0, k0), load_frag_b(w2T, FFD, n0, k0), acc);
#pragma unroll
    for (int r = 0; r < 8; ++r) {
      int row = row0 + part * 8 + r, col = n0 + c;
      hmat[(size_t)row * DMODEL + col] += acc[r] + b2[col];
    }
  }
}

// emb = h[:, -1, :]; s = relu(emb@wn1+bn1)@wn2+bn2; softmax over assets per batch
__global__ void k_head(const float* __restrict__ h_all, const float* __restrict__ wn1,
                       const float* __restrict__ bn1, const float* __restrict__ wn2,
                       const float* __restrict__ bn2, float* __restrict__ out) {
  __shared__ float midl[64 * 16];
  __shared__ float sv[64];
  __shared__ float stats[2];
  int b = blockIdx.x, tid = threadIdx.x;
  for (int idx = tid; idx < 64 * 16; idx += blockDim.x) {
    int a = idx >> 4, j = idx & 15;
    const float* emb = h_all + (((size_t)(b * 64 + a) * LSEQ) + (LSEQ - 1)) * DMODEL;
    float acc = bn1[j];
    for (int dd = 0; dd < DMODEL; ++dd) acc += emb[dd] * wn1[dd * 16 + j];
    midl[idx] = fmaxf(acc, 0.f);
  }
  __syncthreads();
  if (tid < 64) {
    float acc = bn2[0];
#pragma unroll
    for (int j = 0; j < 16; ++j) acc += midl[tid * 16 + j] * wn2[j];
    sv[tid] = acc;
  }
  __syncthreads();
  if (tid == 0) {
    float m = -1e30f;
    for (int a = 0; a < 64; ++a) m = fmaxf(m, sv[a]);
    float s = 0.f;
    for (int a = 0; a < 64; ++a) s += expf(sv[a] - m);
    stats[0] = m; stats[1] = s;
  }
  __syncthreads();
  if (tid < 64) out[b * 64 + tid] = expf(sv[tid] - stats[0]) / stats[1];
}

// ---------------- launch ----------------

extern "C" void kernel_launch(void* const* d_in, const int* in_sizes, int n_in,
                              void* d_out, int out_size, void* d_ws, size_t ws_size,
                              hipStream_t stream) {
  (void)in_sizes; (void)n_in; (void)out_size; (void)ws_size;
  const float* x     = (const float*)d_in[0];
  const float* w_in  = (const float*)d_in[1];
  const float* b_in  = (const float*)d_in[2];
  const float* ln1_g = (const float*)d_in[3];
  const float* ln1_b = (const float*)d_in[4];
  const float* wq    = (const float*)d_in[5];
  const float* wk    = (const float*)d_in[6];
  const float* wv    = (const float*)d_in[7];
  const float* wo    = (const float*)d_in[8];
  const float* bo    = (const float*)d_in[9];
  const float* proj  = (const float*)d_in[10];
  const float* ln2_g = (const float*)d_in[11];
  const float* ln2_b = (const float*)d_in[12];
  const float* w1    = (const float*)d_in[13];
  const float* b1    = (const float*)d_in[14];
  const float* w2    = (const float*)d_in[15];
  const float* b2    = (const float*)d_in[16];
  const float* wn1   = (const float*)d_in[17];
  const float* bn1   = (const float*)d_in[18];
  const float* wn2   = (const float*)d_in[19];
  const float* bn2   = (const float*)d_in[20];

  char* ws = (char*)d_ws;
  float* h = (float*)ws;                        // 64 MB residual stream (f32)
  h16* hn  = (h16*)(ws + (64ull  << 20));       // 32 MB LN output (f16)
  h16* ob  = (h16*)(ws + (96ull  << 20));       // 32 MB attention output (f16)
  h16* wT  = (h16*)(ws + (128ull << 20));       // f16 transposed weights

  const int ATTN_SMEM = 261888;                 // < 320 KB WGP LDS
  (void)hipFuncSetAttribute((const void*)k_attention,
                            hipFuncAttributeMaxDynamicSharedMemorySize, ATTN_SMEM);

  k_prep_weights<<<96, 256, 0, stream>>>(wq, wk, wv, wo, w1, w2, wT);
  k_input_proj<<<SEQ, 256, 0, stream>>>(x, w_in, b_in, h);
  for (int l = 0; l < 2; ++l) {
    const h16* wTl  = wT + (size_t)l * WL;
    const float* pm = proj + (size_t)l * MFEAT * DHEAD;
    k_layernorm<<<SEQ * LSEQ / 8, 256, 0, stream>>>(h, ln1_g + l * DMODEL,
                                                    ln1_b + l * DMODEL, hn);
    k_attention<<<SEQ, 128, ATTN_SMEM, stream>>>(hn, wTl, pm, ob);
    k_wo_residual<<<SEQ * 32, 128, 0, stream>>>(ob, wTl + WOT, bo + l * DMODEL, h);
    k_layernorm<<<SEQ * LSEQ / 8, 256, 0, stream>>>(h, ln2_g + l * DMODEL,
                                                    ln2_b + l * DMODEL, hn);
    k_ffn<<<SEQ * 32, 128, 0, stream>>>(hn, wTl + W1T, wTl + W2T,
                                        b1 + l * FFD, b2 + l * DMODEL, h);
  }
  k_head<<<8, 256, 0, stream>>>(h, wn1, bn1, wn2, bn2, (float*)d_out);
}
